// KPositiveLossCLIP_80874234184251
// MI455X (gfx1250) — matrix operand order repaired and checked by reading, not compiled, not run on hardware
//
#include <hip/hip_runtime.h>
#include <hip/hip_bf16.h>

#define B_SZ 4096
#define D_SZ 128
#define N_SZ 8192
#define K_POS 6
#define INV_T 10.0f

typedef __attribute__((ext_vector_type(16))) __bf16 v16bf;
typedef __attribute__((ext_vector_type(8)))  __bf16 v8bf;
typedef __attribute__((ext_vector_type(8)))  float  v8f;

union V16U { v16bf v; v8bf h[2]; };

// ---------------------------------------------------------------------------
// 1) L2-normalize rows of out0/out1 and pack to bf16 (row-major [B][D]).
// ---------------------------------------------------------------------------
__global__ __launch_bounds__(128) void normalize_pack(
    const float* __restrict__ out0, const float* __restrict__ out1,
    __bf16* __restrict__ An, __bf16* __restrict__ Bn) {
  __shared__ float red[128];
  int row = blockIdx.x;
  const float* src;
  __bf16* dst;
  if (row < B_SZ) { src = out0 + (size_t)row * D_SZ; dst = An + (size_t)row * D_SZ; }
  else            { row -= B_SZ; src = out1 + (size_t)row * D_SZ; dst = Bn + (size_t)row * D_SZ; }
  const int t = threadIdx.x;
  float x = src[t];
  red[t] = x * x;
  __syncthreads();
  for (int off = 64; off > 0; off >>= 1) {
    if (t < off) red[t] += red[t + off];
    __syncthreads();
  }
  float n = fmaxf(sqrtf(red[0]), 1e-12f);
  dst[t] = (__bf16)(x / n);
}

// ---------------------------------------------------------------------------
// 2) logits01 = An @ Bn^T, block-tiled 128(M) x 64(N) through LDS.
//    A/B tiles staged with GLOBAL_LOAD_ASYNC_TO_LDS_B128 (ASYNCcnt path),
//    8 waves x 16 v_wmma_f32_16x16x32_bf16 (K = 128 fully resident).
// ---------------------------------------------------------------------------
#define A_LDS_STRIDE 136   // 128 + 8 bf16 pad (16B) -> breaks 256B bank pattern
__global__ __launch_bounds__(256) void gemm_bf16_wmma(
    const __bf16* __restrict__ A, const __bf16* __restrict__ Bm,
    float* __restrict__ C) {
  __shared__ __bf16 Alds[128 * A_LDS_STRIDE];  // 34816 B
  __shared__ __bf16 Blds[64  * A_LDS_STRIDE];  // 17408 B

  const int tid = threadIdx.x;
  const int tm0 = (blockIdx.x >> 6) * 128;   // 32 M-blocks
  const int tn0 = (blockIdx.x & 63) * 64;    // 64 N-blocks

  // ---- async stage: A tile = 2048 16B-chunks, B tile = 1024 16B-chunks ----
#pragma unroll
  for (int i = 0; i < 8; ++i) {
    const int chunk = tid + i * 256;
    const int row = chunk >> 4;
    const int c16 = chunk & 15;
    const __bf16* g = A + (size_t)(tm0 + row) * D_SZ + c16 * 8;
    unsigned l = (unsigned)(size_t)&Alds[row * A_LDS_STRIDE + c16 * 8];
    asm volatile("global_load_async_to_lds_b128 %0, %1, off"
                 :: "v"(l), "v"(g) : "memory");
  }
#pragma unroll
  for (int i = 0; i < 4; ++i) {
    const int chunk = tid + i * 256;
    const int row = chunk >> 4;
    const int c16 = chunk & 15;
    const __bf16* g = Bm + (size_t)(tn0 + row) * D_SZ + c16 * 8;
    unsigned l = (unsigned)(size_t)&Blds[row * A_LDS_STRIDE + c16 * 8];
    asm volatile("global_load_async_to_lds_b128 %0, %1, off"
                 :: "v"(l), "v"(g) : "memory");
  }
  asm volatile("s_wait_asynccnt 0" ::: "memory");
  __syncthreads();

  // ---- compute: wave w -> rows [w*16, w*16+16), all 4 N-tiles ----
  const int wave = tid >> 5;
  const int lane = tid & 31;
  const int half = lane >> 4;
  const int lrow = lane & 15;

  // A fragments (16-bit A 16x32 layout): lanes 0-15 K{0..7,16..23}, 16-31 K{8..15,24..31}
  V16U af[4];
  const int arow = wave * 16 + lrow;
#pragma unroll
  for (int kt = 0; kt < 4; ++kt) {
    const __bf16* p = &Alds[arow * A_LDS_STRIDE + kt * 32 + half * 8];
    af[kt].h[0] = *reinterpret_cast<const v8bf*>(p);
    af[kt].h[1] = *reinterpret_cast<const v8bf*>(p + 16);
  }

  float* cbase = C + (size_t)(tm0 + wave * 16 + 8 * half) * B_SZ + tn0 + lrow;
#pragma unroll
  for (int nt = 0; nt < 4; ++nt) {
    v8f acc = {};
#pragma unroll
    for (int kt = 0; kt < 4; ++kt) {
      // B fragments (16-bit B 32x16 layout): lanes 0-15 K{0..7,16..23}, 16-31 K{8..15,24..31}
      const __bf16* p = &Blds[(nt * 16 + lrow) * A_LDS_STRIDE + kt * 32 + half * 8];
      V16U bf_;
      bf_.h[0] = *reinterpret_cast<const v8bf*>(p);
      bf_.h[1] = *reinterpret_cast<const v8bf*>(p + 16);
      acc = __builtin_amdgcn_wmma_f32_16x16x32_bf16(
          /*neg_a=*/false, af[kt].v, /*neg_b=*/false, bf_.v,
          /*c_mod=*/(short)0, acc, /*reuse_a=*/false, /*reuse_b=*/false);
    }
    // C/D layout: VGPR r, lane l -> M = r + 8*(l>=16), N = l&15
    float* crow = cbase + nt * 16;
#pragma unroll
    for (int r = 0; r < 8; ++r) crow[(size_t)r * B_SZ] = acc[r];
  }
}

// ---------------------------------------------------------------------------
// 3) nnz_i = #nonzero of sim[indices[i], indices[:]] ; 4 rows per block,
//    indices cached in LDS, integer shared atomics (deterministic counts).
// ---------------------------------------------------------------------------
__global__ __launch_bounds__(256) void nnz_count(
    const int* __restrict__ idx, const float* __restrict__ sim,
    int* __restrict__ nnz) {
  __shared__ int sidx[B_SZ];   // 16 KB
  __shared__ int cnt[4];
  for (int j = threadIdx.x; j < B_SZ; j += 256) sidx[j] = idx[j];
  if (threadIdx.x < 4) cnt[threadIdx.x] = 0;
  __syncthreads();
  for (int r = 0; r < 4; ++r) {
    const int i = blockIdx.x * 4 + r;
    const float* srow = sim + (size_t)sidx[i] * N_SZ;
    int local = 0;
    for (int j = threadIdx.x; j < B_SZ; j += 256)
      local += (srow[sidx[j]] != 0.0f) ? 1 : 0;
    atomicAdd(&cnt[r], local);
    __syncthreads();
  }
  if (threadIdx.x < 4) nnz[blockIdx.x * 4 + threadIdx.x] = cnt[threadIdx.x];
}

// ---------------------------------------------------------------------------
// 4) Deterministic positive-column sampling: cols[i][0]=i, k-1 hashed ints
//    in [0, nnz_i)  (jax threefry not replicable; compile-only loop).
// ---------------------------------------------------------------------------
__device__ __forceinline__ unsigned hash_u32(unsigned x) {
  x ^= x >> 16; x *= 0x7feb352dU; x ^= x >> 15; x *= 0x846ca68bU; x ^= x >> 16;
  return x;
}
__global__ void sample_cols(const int* __restrict__ nnz, int* __restrict__ cols) {
  const int i = blockIdx.x * blockDim.x + threadIdx.x;
  if (i >= B_SZ) return;
  int n = nnz[i]; if (n < 1) n = 1;
  cols[i * K_POS] = i;
  for (int k = 1; k < K_POS; ++k)
    cols[i * K_POS + k] =
        (int)(hash_u32(0x9E3779B9u ^ (unsigned)(i * 131 + k)) % (unsigned)n);
}

// ---------------------------------------------------------------------------
// 5) Row LSE: one block per row, online (m,s) per thread + LDS merge.
// ---------------------------------------------------------------------------
__global__ __launch_bounds__(256) void row_lse(
    const float* __restrict__ L, float* __restrict__ lse) {
  __shared__ float sm[256], ss[256];
  const float* row = L + (size_t)blockIdx.x * B_SZ;
  float m = -INFINITY, s = 0.f;
  for (int j = threadIdx.x; j < B_SZ; j += 256) {
    float x = row[j] * INV_T;
    float mo = m; m = fmaxf(m, x);
    s = s * __expf(mo - m) + __expf(x - m);
  }
  sm[threadIdx.x] = m; ss[threadIdx.x] = s;
  __syncthreads();
  for (int off = 128; off > 0; off >>= 1) {
    if (threadIdx.x < off) {
      float ma = sm[threadIdx.x], mb = sm[threadIdx.x + off];
      float m2 = fmaxf(ma, mb);
      ss[threadIdx.x] = ss[threadIdx.x] * __expf(ma - m2) +
                        ss[threadIdx.x + off] * __expf(mb - m2);
      sm[threadIdx.x] = m2;
    }
    __syncthreads();
  }
  if (threadIdx.x == 0) lse[blockIdx.x] = sm[0] + __logf(ss[0]);
}

// ---------------------------------------------------------------------------
// 6) Column LSE (transposed half of logits), coalesced split-K:
//    grid (16 col-chunks, 32 row-slices of 128); partial (m,s) per column.
// ---------------------------------------------------------------------------
__global__ __launch_bounds__(256) void col_partial(
    const float* __restrict__ L, float2* __restrict__ part) {
  const int c  = blockIdx.x * 256 + threadIdx.x;
  const int r0 = blockIdx.y * 128;
  float m = -INFINITY, s = 0.f;
  for (int r = r0; r < r0 + 128; ++r) {
    float x = L[(size_t)r * B_SZ + c] * INV_T;
    float mo = m; m = fmaxf(m, x);
    s = s * __expf(mo - m) + __expf(x - m);
  }
  part[(size_t)blockIdx.y * B_SZ + c] = make_float2(m, s);
}

__global__ void col_combine(const float2* __restrict__ part,
                            float* __restrict__ lse) {
  const int c = blockIdx.x * blockDim.x + threadIdx.x;
  if (c >= B_SZ) return;
  float m = -INFINITY, s = 0.f;
  for (int y = 0; y < 32; ++y) {
    float2 p = part[(size_t)y * B_SZ + c];
    float m2 = fmaxf(m, p.x);
    s = s * __expf(m - m2) + p.y * __expf(p.x - m2);
    m = m2;
  }
  lse[c] = m + __logf(s);
}

// ---------------------------------------------------------------------------
// 7) Per-row loss contribution (top row i + transposed row B+i), with dedupe
//    of sampled columns (scatter .set(1) semantics).
// ---------------------------------------------------------------------------
__global__ void loss_per_row(const float* __restrict__ L,
                             const int* __restrict__ cols,
                             const float* __restrict__ lseR,
                             const float* __restrict__ lseC,
                             float* __restrict__ part) {
  const int i = blockIdx.x * blockDim.x + threadIdx.x;
  if (i >= B_SZ) return;
  int c[K_POS];
#pragma unroll
  for (int k = 0; k < K_POS; ++k) c[k] = cols[i * K_POS + k];
  float sumTop = 0.f, sumBot = 0.f;
  int npos = 0;
#pragma unroll
  for (int k = 0; k < K_POS; ++k) {
    bool dup = false;
    for (int p = 0; p < k; ++p) dup |= (c[p] == c[k]);
    if (dup) continue;
    ++npos;
    sumTop += L[(size_t)i * B_SZ + c[k]] * INV_T;
    sumBot += L[(size_t)c[k] * B_SZ + i] * INV_T;
  }
  const float top = sumTop - (float)npos * lseR[i];
  const float bot = sumBot - (float)npos * lseC[i];
  part[i] = -(top + bot);
}

__global__ __launch_bounds__(1024) void reduce_loss(
    const float* __restrict__ part, float* __restrict__ out) {
  __shared__ float red[1024];
  float s = 0.f;
  for (int i = threadIdx.x; i < B_SZ; i += 1024) s += part[i];
  red[threadIdx.x] = s;
  __syncthreads();
  for (int off = 512; off > 0; off >>= 1) {
    if (threadIdx.x < off) red[threadIdx.x] += red[threadIdx.x + off];
    __syncthreads();
  }
  if (threadIdx.x == 0)
    out[0] = red[0] / ((float)B_SZ * (float)(K_POS - 1)) / (2.0f * (float)B_SZ);
}

__global__ void arange_fill(float* __restrict__ out) {
  const int i = blockIdx.x * blockDim.x + threadIdx.x;
  if (i < B_SZ) out[i] = (float)i;
}

// ---------------------------------------------------------------------------
extern "C" void kernel_launch(void* const* d_in, const int* in_sizes, int n_in,
                              void* d_out, int out_size, void* d_ws, size_t ws_size,
                              hipStream_t stream) {
  (void)in_sizes; (void)n_in; (void)out_size; (void)ws_size;
  const float* out0    = (const float*)d_in[0];
  const float* out1    = (const float*)d_in[1];
  const int*   indices = (const int*)d_in[2];
  const float* sim     = (const float*)d_in[3];

  float* out    = (float*)d_out;
  float* loss   = out;                          // [1]
  float* logits = out + 1;                      // [B*B]
  float* ar     = out + 1 + (size_t)B_SZ * B_SZ;// [B]

  char* ws = (char*)d_ws;
  __bf16* An   = (__bf16*)(ws);                                  // 1 MB
  __bf16* Bn   = (__bf16*)(ws + (1u << 20));                     // 1 MB
  int*    nnz  = (int*)  (ws + (2u << 20));                      // 16 KB
  int*    cols = (int*)  (ws + (2u << 20) + (64u << 10));        // 96 KB
  float*  lseR = (float*)(ws + (2u << 20) + (256u << 10));       // 16 KB
  float*  lseC = (float*)(ws + (2u << 20) + (272u << 10));       // 16 KB
  float2* cprt = (float2*)(ws + (2u << 20) + (512u << 10));      // 1 MB
  float*  lprt = (float*)(ws + (3u << 20) + (512u << 10));       // 16 KB

  normalize_pack<<<2 * B_SZ, 128, 0, stream>>>(out0, out1, An, Bn);
  // 32 M-blocks x 64 N-blocks of 128x64 tiles
  gemm_bf16_wmma<<<2048, 256, 0, stream>>>(An, Bn, logits);
  nnz_count<<<B_SZ / 4, 256, 0, stream>>>(indices, sim, nnz);
  sample_cols<<<B_SZ / 256, 256, 0, stream>>>(nnz, cols);
  row_lse<<<B_SZ, 256, 0, stream>>>(logits, lseR);
  dim3 gcp(16, 32);
  col_partial<<<gcp, 256, 0, stream>>>(logits, cprt);
  col_combine<<<B_SZ / 256, 256, 0, stream>>>(cprt, lseC);
  loss_per_row<<<B_SZ / 256, 256, 0, stream>>>(logits, cols, lseR, lseC, lprt);
  reduce_loss<<<1, 1024, 0, stream>>>(lprt, loss);
  arange_fill<<<B_SZ / 256, 256, 0, stream>>>(ar);
}
